// LSTM_51316269252991
// MI455X (gfx1250) — matrix operand order repaired and checked by reading, not compiled, not run on hardware
//
#include <hip/hip_runtime.h>

// ---------------------------------------------------------------------------
// Fused 3-layer LSTM (B=1024, T=2048, H=32) + final FC, CDNA5 / gfx1250.
//
// One wave32 per 16-row batch tile runs the whole time loop for all three
// layers. Gate pre-activations computed transposed:
//     G^T[128 x 16] = W_ih[128 x K] * in^T[K x 16] + W_hh[128 x 32] * h^T + b
// as 8 accumulator tiles via v_wmma_f32_16x16x32_f16 (K=32; layer0 K padded
// 12->32 with zeros in the A operand, which makes the B operand's padding
// lanes/elements don't-care). Batch column lives in the same lane for the B
// operand and the C/D accumulator, so h(C-layout,f32) -> B-operand(f16) is a
// lane^16 shuffle + cvt -- no LDS in the recurrence. Weights stay resident in
// VGPRs (>256 regs via VGPR-MSB banks). Nonlinearities use v_exp_f32 +
// v_rcp_f32 (TRANS ops co-execute with XDL WMMA). x loads are software-
// pipelined one timestep ahead.
// ---------------------------------------------------------------------------

typedef __attribute__((ext_vector_type(16))) _Float16 v16h;
typedef __attribute__((ext_vector_type(8)))  float    v8f;

#define B_   1024
#define T_   2048
#define DIN  12
#define H_   32
#define G4   128   // 4*H
#define OUTD 4

// fast sigmoid / tanh: v_exp_f32 + v_rcp_f32, no IEEE-div expansion
__device__ __forceinline__ float sigm_f(float x) {
    return __builtin_amdgcn_rcpf(1.0f + __expf(-x));
}
__device__ __forceinline__ float tanh_f(float x) {
    // tanh(x) = 1 - 2/(exp(2x)+1)
    return __builtin_fmaf(-2.0f, __builtin_amdgcn_rcpf(__expf(2.0f * x) + 1.0f), 1.0f);
}

__launch_bounds__(32, 1)
__global__ void lstm3_fused_wmma(
    const float* __restrict__ x,
    const float* __restrict__ Wih0, const float* __restrict__ Whh0,
    const float* __restrict__ bih0, const float* __restrict__ bhh0,
    const float* __restrict__ Wih1, const float* __restrict__ Whh1,
    const float* __restrict__ bih1, const float* __restrict__ bhh1,
    const float* __restrict__ Wih2, const float* __restrict__ Whh2,
    const float* __restrict__ bih2, const float* __restrict__ bhh2,
    const float* __restrict__ fcw,  const float* __restrict__ fcb,
    float* __restrict__ out)
{
    __shared__ __align__(32) float lds_bc[3][G4];   // combined biases b_ih + b_hh
    __shared__ float lds_h[H_ * 16];                // final-step h transpose buffer

    const int lane = threadIdx.x;        // 0..31
    const int r    = lane & 15;          // row/col within 16-tile
    const int hi   = lane >> 4;          // lane half (0/1)
    const int base = blockIdx.x * 16;    // batch tile origin

    // ---- combined biases into LDS (once) ----------------------------------
    const float* bi_[3] = { bih0, bih1, bih2 };
    const float* bh_[3] = { bhh0, bhh1, bhh2 };
    #pragma unroll
    for (int l = 0; l < 3; ++l)
        for (int k = lane; k < G4; k += 32)
            lds_bc[l][k] = bi_[l][k] + bh_[l][k];
    __syncthreads();

    // ---- gather weights into registers as WMMA A-layout f16 tiles ---------
    // A tile m of W[128][din]: lane holds row M = 16m + r,
    // element e maps to K = hi*16 + e   (contiguous k order, matches B fills)
    const float* Wih_[3] = { Wih0, Wih1, Wih2 };
    const float* Whh_[3] = { Whh0, Whh1, Whh2 };
    const int    din_[3] = { DIN, H_, H_ };

    v16h a_ih[3][8], a_hh[3][8];
    #pragma unroll
    for (int l = 0; l < 3; ++l) {
        #pragma unroll
        for (int m = 0; m < 8; ++m) {
            const float* wi = Wih_[l] + (16 * m + r) * din_[l];
            const float* wh = Whh_[l] + (16 * m + r) * H_;
            v16h va, vh;
            #pragma unroll
            for (int e = 0; e < 16; ++e) {
                const int K = hi * 16 + e;
                va[e] = (K < din_[l]) ? (_Float16)wi[K] : (_Float16)0.0f;
                vh[e] = (_Float16)wh[K];
            }
            a_ih[l][m] = va;
            a_hh[l][m] = vh;
        }
    }

    // ---- recurrent state: c in C-layout (f32), h as B-operand (f16) -------
    v8f  cst[3][2];
    v16h hB[3];
    #pragma unroll
    for (int l = 0; l < 3; ++l) {
        #pragma unroll
        for (int j = 0; j < 2; ++j)
            #pragma unroll
            for (int v = 0; v < 8; ++v) cst[l][j][v] = 0.0f;
        #pragma unroll
        for (int e = 0; e < 16; ++e) hB[l][e] = (_Float16)0.0f;
    }

    v8f hf0, hf1;   // layer-2 h at final step (C layout)
    #pragma unroll
    for (int v = 0; v < 8; ++v) { hf0[v] = 0.0f; hf1[v] = 0.0f; }

    // Every lane loads batch row r (hi-lanes duplicate their partner's row --
    // their B elements hit zero columns of the layer-0 A tiles, so values are
    // don't-care as long as they're finite). No EXEC divergence in the loop.
    const float* xrow = x + ((long)(base + r) * T_) * DIN;

    // software pipeline: prefetch x for t=0
    float4 p0 = *(const float4*)(xrow);
    float4 p1 = *(const float4*)(xrow + 4);
    float4 p2 = *(const float4*)(xrow + 8);

    // ======================= time loop =====================================
    #pragma unroll 1
    for (int t = 0; t < T_; ++t) {
        const float4 c0 = p0, c1 = p1, c2 = p2;
        // prefetch next step (clamped: t=T-1 reloads the same row, in-bounds)
        {
            const int tn = (t + 1 < T_) ? (t + 1) : (T_ - 1);
            const float* xp = xrow + (long)tn * DIN;
            p0 = *(const float4*)(xp);
            p1 = *(const float4*)(xp + 4);
            p2 = *(const float4*)(xp + 8);
        }

        v16h xB;
        xB[0]  = (_Float16)c0.x;  xB[1]  = (_Float16)c0.y;
        xB[2]  = (_Float16)c0.z;  xB[3]  = (_Float16)c0.w;
        xB[4]  = (_Float16)c1.x;  xB[5]  = (_Float16)c1.y;
        xB[6]  = (_Float16)c1.z;  xB[7]  = (_Float16)c1.w;
        xB[8]  = (_Float16)c2.x;  xB[9]  = (_Float16)c2.y;
        xB[10] = (_Float16)c2.z;  xB[11] = (_Float16)c2.w;
        xB[12] = (_Float16)0.0f;  xB[13] = (_Float16)0.0f;
        xB[14] = (_Float16)0.0f;  xB[15] = (_Float16)0.0f;

        v16h inB = xB;
        #pragma unroll
        for (int l = 0; l < 3; ++l) {
            // G^T tiles: m=0,1 -> i ; 2,3 -> f ; 4,5 -> g ; 6,7 -> o
            v8f acc[8];
            // pass 1: bias seed + input contribution (8 independent WMMAs)
            #pragma unroll
            for (int m = 0; m < 8; ++m) {
                const float* bp = &lds_bc[l][16 * m + 8 * hi];
                v8f a;
                #pragma unroll
                for (int v = 0; v < 8; ++v) a[v] = bp[v];
                acc[m] = __builtin_amdgcn_wmma_f32_16x16x32_f16(
                             false, a_ih[l][m], false, inB, (short)0, a, false, false);
            }
            // pass 2: recurrent contribution (8 independent WMMAs)
            #pragma unroll
            for (int m = 0; m < 8; ++m) {
                acc[m] = __builtin_amdgcn_wmma_f32_16x16x32_f16(
                             false, a_hh[l][m], false, hB[l], (short)0, acc[m], false, false);
            }

            // elementwise LSTM cell update (C layout, hidden half j)
            v8f h0, h1;
            #pragma unroll
            for (int j = 0; j < 2; ++j) {
                v8f hc;
                #pragma unroll
                for (int v = 0; v < 8; ++v) {
                    const float iv = sigm_f(acc[0 + j][v]);
                    const float fv = sigm_f(acc[2 + j][v]);
                    const float gv = tanh_f(acc[4 + j][v]);
                    const float ov = sigm_f(acc[6 + j][v]);
                    const float cv = fv * cst[l][j][v] + iv * gv;
                    cst[l][j][v] = cv;
                    hc[v] = ov * tanh_f(cv);
                }
                if (j == 0) h0 = hc; else h1 = hc;
            }
            if (l == 2) { hf0 = h0; hf1 = h1; }

            // h (C layout f32) -> B operand (f16): lane^16 exchange, no LDS.
            // low lanes:  e[v]=h0[v],         e[8+v]=partner h0[v]
            // high lanes: e[v]=partner h1[v], e[8+v]=h1[v]
            v16h nb;
            #pragma unroll
            for (int v = 0; v < 8; ++v) {
                const float q0 = __shfl_xor(h0[v], 16, 32);
                const float q1 = __shfl_xor(h1[v], 16, 32);
                const float elo = hi ? q1 : h0[v];
                const float ehi = hi ? h1[v] : q0;
                nb[v]     = (_Float16)elo;
                nb[8 + v] = (_Float16)ehi;
            }
            hB[l] = nb;
            inB   = nb;   // feed next layer at same timestep
        }
    }

    // ---- final FC on last-step h of layer 2 -------------------------------
    // C element v of tile j: hidden = 16j + 8*hi + v, batch col = r
    #pragma unroll
    for (int v = 0; v < 8; ++v) {
        lds_h[(0  + 8 * hi + v) * 16 + r] = hf0[v];
        lds_h[(16 + 8 * hi + v) * 16 + r] = hf1[v];
    }
    __syncthreads();
    if (hi == 0) {
        float o0 = fcb[0], o1 = fcb[1], o2 = fcb[2], o3 = fcb[3];
        #pragma unroll
        for (int k = 0; k < H_; ++k) {
            const float hv = lds_h[k * 16 + r];
            o0 += hv * fcw[0 * H_ + k];
            o1 += hv * fcw[1 * H_ + k];
            o2 += hv * fcw[2 * H_ + k];
            o3 += hv * fcw[3 * H_ + k];
        }
        float* op = out + (long)(base + r) * OUTD;
        op[0] = o0; op[1] = o1; op[2] = o2; op[3] = o3;
    }
}

extern "C" void kernel_launch(void* const* d_in, const int* in_sizes, int n_in,
                              void* d_out, int out_size, void* d_ws, size_t ws_size,
                              hipStream_t stream) {
    const float* x    = (const float*)d_in[0];
    const float* Wih0 = (const float*)d_in[1];
    const float* Whh0 = (const float*)d_in[2];
    const float* bih0 = (const float*)d_in[3];
    const float* bhh0 = (const float*)d_in[4];
    const float* Wih1 = (const float*)d_in[5];
    const float* Whh1 = (const float*)d_in[6];
    const float* bih1 = (const float*)d_in[7];
    const float* bhh1 = (const float*)d_in[8];
    const float* Wih2 = (const float*)d_in[9];
    const float* Whh2 = (const float*)d_in[10];
    const float* bih2 = (const float*)d_in[11];
    const float* bhh2 = (const float*)d_in[12];
    const float* fcw  = (const float*)d_in[13];
    const float* fcb  = (const float*)d_in[14];
    float* out = (float*)d_out;

    // 1024 batch rows / 16 per wave = 64 workgroups of one wave32 each.
    lstm3_fused_wmma<<<dim3(B_ / 16), dim3(32), 0, stream>>>(
        x, Wih0, Whh0, bih0, bhh0, Wih1, Whh1, bih1, bhh1,
        Wih2, Whh2, bih2, bhh2, fcw, fcb, out);
}
